// ResFCEventuallyLayer_82205674045459
// MI455X (gfx1250) — compile-verified, matrix operand order checked by reading, not run on hardware
//
#include <hip/hip_runtime.h>

// Problem constants (match the reference).
#define Bn 4096   // batch rows (M)
#define Nn 4096   // output cols (N)
#define Kc 4096   // contraction (K)

typedef __attribute__((ext_vector_type(16))) __bf16 v16bf;
typedef __attribute__((ext_vector_type(8)))  float  v8f;

// Round-to-nearest-even f32 -> bf16, packed pairwise into a dword.
static __device__ __forceinline__ unsigned int pack2_bf16(float a, float b) {
  unsigned int ua = __float_as_uint(a);
  unsigned int ub = __float_as_uint(b);
  ua += 0x7FFFu + ((ua >> 16) & 1u);
  ub += 0x7FFFu + ((ub >> 16) & 1u);
  return (ua >> 16) | (ub & 0xFFFF0000u);
}

// One-shot conversion pass: f32 x / W -> bf16 copies in workspace.
// 8 elements per lane: 2x float4 in, 1x uint4 out (b128 traffic both ways).
__global__ __launch_bounds__(256) void cvt_bf16_kernel(
    const float* __restrict__ x, const float* __restrict__ w,
    unsigned short* __restrict__ xbf, unsigned short* __restrict__ wbf) {
  const float* src = blockIdx.y ? w : x;
  unsigned short* dst = blockIdx.y ? wbf : xbf;
  size_t i = ((size_t)blockIdx.x * blockDim.x + threadIdx.x) * 8;
  float4 f0 = *(const float4*)(src + i);
  float4 f1 = *(const float4*)(src + i + 4);
  uint4 o;
  o.x = pack2_bf16(f0.x, f0.y);
  o.y = pack2_bf16(f0.z, f0.w);
  o.z = pack2_bf16(f1.x, f1.y);
  o.w = pack2_bf16(f1.z, f1.w);
  *(uint4*)(dst + i) = o;
}

union FragU { v16bf v; uint4 u[2]; };

// NT bf16 WMMA GEMM with fused ResFCEventually epilogue.
// Block: 256 threads (8 wave32s), wave grid 2(M) x 4(N), wave tile 64x64,
// block tile 128(M) x 256(N). K stepped 32 per v_wmma_f32_16x16x32_bf16.
__global__ __launch_bounds__(256) void gemm_resfc_kernel(
    const unsigned short* __restrict__ xbf,   // [Bn,Kc] bf16 row-major
    const unsigned short* __restrict__ wbf,   // [Nn,Kc] bf16 row-major
    const float* __restrict__ x,              // original f32 x for epilogue
    const float* __restrict__ betap,          // scalar beta
    const float* __restrict__ beta_res,       // [Nn]
    float* __restrict__ out) {                // [Bn,Nn]
  const int lane = threadIdx.x & 31;
  const int wid  = threadIdx.x >> 5;
  const int wm   = wid & 1;   // 2 waves along M
  const int wn   = wid >> 1;  // 4 waves along N
  const int rowBase = blockIdx.y * 128 + wm * 64;
  const int colBase = blockIdx.x * 256 + wn * 64;
  const int lr = lane & 15;            // row-within-16 for A/B operands
  const int kh = (lane >> 4) * 8;      // K-half element offset (0 or 8)

  // Per-lane base pointers: A lane holds x[row, k..], B lane holds W[col, k..]
  // (identical addressing: both operands contract over their contiguous axis).
  const unsigned short* aP[4];
  const unsigned short* bP[4];
#pragma unroll
  for (int i = 0; i < 4; ++i) {
    aP[i] = xbf + (size_t)(rowBase + i * 16 + lr) * Kc + kh;
    bP[i] = wbf + (size_t)(colBase + i * 16 + lr) * Kc + kh;
  }

  const v8f vzero = {0.f, 0.f, 0.f, 0.f, 0.f, 0.f, 0.f, 0.f};
  v8f acc[4][4];
#pragma unroll
  for (int i = 0; i < 4; ++i)
#pragma unroll
    for (int j = 0; j < 4; ++j) acc[i][j] = vzero;

  for (int k0 = 0; k0 < Kc; k0 += 32) {
    FragU a[4], b[4];
#pragma unroll
    for (int i = 0; i < 4; ++i) {
      // Fragment = two b128 loads: K-chunk at +0 and +16 elements (ISA layout:
      // lanes 0-15 carry K 0..7 / 16..23, lanes 16-31 carry K 8..15 / 24..31).
      const uint4* p = (const uint4*)(aP[i] + k0);
      a[i].u[0] = p[0];
      a[i].u[1] = p[2];
      const uint4* q = (const uint4*)(bP[i] + k0);
      b[i].u[0] = q[0];
      b[i].u[1] = q[2];
    }
#pragma unroll
    for (int i = 0; i < 4; ++i)
#pragma unroll
      for (int j = 0; j < 4; ++j)
        acc[i][j] = __builtin_amdgcn_wmma_f32_16x16x32_bf16(
            false, a[i].v, false, b[j].v, (short)0, acc[i][j], false, false);
  }

  // Fused epilogue. C/D layout: VGPR e, lanes 0-15 -> M=e, N=lane;
  // lanes 16-31 -> M=e+8, N=lane-16.
  const float beta = betap[0];
#pragma unroll
  for (int i = 0; i < 4; ++i) {
    const int m0 = rowBase + i * 16 + (lane >> 4) * 8;
#pragma unroll
    for (int j = 0; j < 4; ++j) {
      const int n = colBase + j * 16 + lr;
      const float br = beta_res[n];
#pragma unroll
      for (int e = 0; e < 8; ++e) {
        const int r = m0 + e;
        const float rs  = acc[i][j][e];
        const float o1  = fmaxf(1.0f - beta + rs, 0.0f);           // hlogic
        const float res = fmaxf(1.0f - (br - (x[(size_t)r * Nn + n] + o1)), 0.0f);
        // LeakyReLU(0.02) after a ReLU is identity, but keep reference form:
        const float fin = res >= 0.0f ? res : 0.02f * res;
        out[(size_t)r * Nn + n] = fin;
      }
    }
  }
}

extern "C" void kernel_launch(void* const* d_in, const int* in_sizes, int n_in,
                              void* d_out, int out_size, void* d_ws, size_t ws_size,
                              hipStream_t stream) {
  const float* x        = (const float*)d_in[0];  // [Bn,Nn]
  const float* W        = (const float*)d_in[1];  // [Nn,Nn]
  const float* beta     = (const float*)d_in[2];  // [1]
  const float* beta_res = (const float*)d_in[3];  // [Nn]
  float* out = (float*)d_out;

  // Workspace: bf16 copies of x and W (32 MB each).
  unsigned short* xbf = (unsigned short*)d_ws;
  unsigned short* wbf = xbf + (size_t)Bn * Kc;

  // 1) f32 -> bf16 conversion (memory-bound, ~24 MB/grid-ms class pass).
  dim3 cgrid((unsigned)(((size_t)Bn * Kc) / (256 * 8)), 2);
  cvt_bf16_kernel<<<cgrid, 256, 0, stream>>>(x, W, xbf, wbf);

  // 2) Fused WMMA GEMM + epilogue.
  dim3 ggrid(Nn / 256, Bn / 128);
  gemm_resfc_kernel<<<ggrid, 256, 0, stream>>>(xbf, wbf, x, beta, beta_res, out);
}
